// CGCNNLayer_15573551415579
// MI455X (gfx1250) — matrix operand order repaired and checked by reading, not compiled, not run on hardware
//
#include <hip/hip_runtime.h>
#include <hip/hip_bf16.h>

#define HID   64
#define ZDIM  192   // 2*HID + EDIM

typedef __attribute__((ext_vector_type(16))) __bf16 v16bf;
typedef __attribute__((ext_vector_type(2)))  __bf16 bf16x2;
typedef __attribute__((ext_vector_type(8)))  float  v8f;
typedef __attribute__((ext_vector_type(4)))  float  f32x4;
typedef __attribute__((ext_vector_type(2)))  float  f32x2;
typedef __attribute__((ext_vector_type(4)))  unsigned int u32x4;

union AFrag { v16bf v; unsigned int u[8]; };
union BFrag { v16bf v; u32x4 q[2]; };

// f32 pair -> packed bf16 pair via vector fptrunc => single v_cvt_pk_bf16_f32
__device__ __forceinline__ unsigned int pk_bf16(float lo, float hi) {
    f32x2 f; f.x = lo; f.y = hi;
    union { bf16x2 b; unsigned int u; } r;
    r.b = __builtin_convertvector(f, bf16x2);
    return r.u;
}
__device__ __forceinline__ unsigned short bf16_bits(float f) {
    union { __bf16 b; unsigned short s; } r;
    r.b = (__bf16)f;
    return r.s;
}

__device__ __forceinline__ void pack8(AFrag& a, f32x4 f0, f32x4 f1, f32x4 f2, f32x4 f3) {
    a.u[0] = pk_bf16(f0.x, f0.y);  a.u[1] = pk_bf16(f0.z, f0.w);
    a.u[2] = pk_bf16(f1.x, f1.y);  a.u[3] = pk_bf16(f1.z, f1.w);
    a.u[4] = pk_bf16(f2.x, f2.y);  a.u[5] = pk_bf16(f2.z, f2.w);
    a.u[6] = pk_bf16(f3.x, f3.y);  a.u[7] = pk_bf16(f3.z, f3.w);
}

// 16 WMMAs for one 32-wide K chunk: 4 N-tiles x 2 projections x 2 M-tiles.
// Each B fragment feeds TWO WMMAs -> halves LDS traffic per FLOP.
__device__ __forceinline__ void mm_step2(
    const AFrag& a0, const AFrag& a1, int kbase, const int* rowOff,
    const unsigned short* sw0, const unsigned short* sw1,
    v8f* sig0, v8f* sig1, v8f* sp0, v8f* sp1)
{
#pragma unroll
    for (int t = 0; t < 4; ++t) {
        const int boff = rowOff[t] + kbase;
        BFrag b0, b1;
        const u32x4* q0 = (const u32x4*)(sw0 + boff);
        const u32x4* q1 = (const u32x4*)(sw1 + boff);
        b0.q[0] = q0[0]; b0.q[1] = q0[2];
        b1.q[0] = q1[0]; b1.q[1] = q1[2];
        sig0[t] = __builtin_amdgcn_wmma_f32_16x16x32_bf16(
            false, a0.v, false, b0.v, (short)0, sig0[t], false, false);
        sig1[t] = __builtin_amdgcn_wmma_f32_16x16x32_bf16(
            false, a1.v, false, b0.v, (short)0, sig1[t], false, false);
        sp0[t]  = __builtin_amdgcn_wmma_f32_16x16x32_bf16(
            false, a0.v, false, b1.v, (short)0, sp0[t], false, false);
        sp1[t]  = __builtin_amdgcn_wmma_f32_16x16x32_bf16(
            false, a1.v, false, b1.v, (short)0, sp1[t], false, false);
    }
}

// ---------------------------------------------------------------------------
// Main edge kernel: 32-edge pairs of 16x16 tiles, dual [32,192]x[192,64]
// bf16 WMMA GEMMs, gated activations, f32 atomic scatter to agg[dst].
// ---------------------------------------------------------------------------
__global__ __launch_bounds__(256) void cgcnn_edge_kernel(
    const float* __restrict__ x, const float* __restrict__ ea,
    const int* __restrict__ src, const int* __restrict__ dst,
    const float* __restrict__ Wsig, const float* __restrict__ bsig,
    const float* __restrict__ Wsp,  const float* __restrict__ bsp,
    float* __restrict__ agg, int nPairs)
{
    // Both weight matrices as bf16 in LDS, plain row-major [64][192] (48 KB).
    __shared__ __align__(16) unsigned short sW[2][HID * ZDIM];
    for (int i = threadIdx.x; i < HID * ZDIM; i += blockDim.x) {
        sW[0][i] = bf16_bits(Wsig[i]);
        sW[1][i] = bf16_bits(Wsp[i]);
    }
    __syncthreads();

    const int lane   = threadIdx.x & 31;
    const int lo16   = lane & 15;          // edge row within tile / N column
    const int half8  = (lane >> 4) << 3;   // 0 or 8  (A layout K sub-offset)
    const int half16 = (lane >> 4) << 4;   // 0 or 16 (unused; B also uses half8)

    const int wavesPerBlk = blockDim.x >> 5;
    const int gWave  = blockIdx.x * wavesPerBlk + (threadIdx.x >> 5);
    const int nWaves = gridDim.x * wavesPerBlk;

    // loop-invariant per-lane values
    int rowOff[4];
    float bsv[4], bpv[4];
#pragma unroll
    for (int t = 0; t < 4; ++t) {
        rowOff[t] = (t * 16 + lo16) * ZDIM + half8;
        bsv[t] = bsig[t * 16 + lo16];
        bpv[t] = bsp [t * 16 + lo16];
    }

    for (int pair = gWave; pair < nPairs; pair += nWaves) {
        const int e0 = pair << 5;
        const int eA = e0 + lo16;          // tile0 edge row for this lane
        const int eB = eA + 16;            // tile1 edge row for this lane
        const int sI0 = src[eA], dI0 = dst[eA];
        const int sI1 = src[eB], dI1 = dst[eB];

        // bases already include the lane's A-layout K sub-offset
        const float* a0s = x  + (size_t)sI0 * HID + half8;
        const float* a0d = x  + (size_t)dI0 * HID + half8;
        const float* a0e = ea + (size_t)eA  * HID + half8;
        const float* a1s = x  + (size_t)sI1 * HID + half8;
        const float* a1d = x  + (size_t)dI1 * HID + half8;
        const float* a1e = ea + (size_t)eB  * HID + half8;

        v8f sig0[4], sig1[4], sp0[4], sp1[4];
#pragma unroll
        for (int t = 0; t < 4; ++t) {
            sig0[t] = (v8f)0.0f; sig1[t] = (v8f)0.0f;
            sp0[t]  = (v8f)0.0f; sp1[t]  = (v8f)0.0f;
        }

        // ---- K chunks 0..3: x[src]/x[dst] gathers (L2-resident, RT) ----
#pragma unroll 1
        for (int c = 0; c < 4; ++c) {
            const float* p0 = ((c & 2) ? a0d : a0s) + (c & 1) * 32;
            const float* p1 = ((c & 2) ? a1d : a1s) + (c & 1) * 32;
            f32x4 f0 = *(const f32x4*)(p0);
            f32x4 f1 = *(const f32x4*)(p0 + 4);
            f32x4 f2 = *(const f32x4*)(p0 + 16);
            f32x4 f3 = *(const f32x4*)(p0 + 20);
            f32x4 g0 = *(const f32x4*)(p1);
            f32x4 g1 = *(const f32x4*)(p1 + 4);
            f32x4 g2 = *(const f32x4*)(p1 + 16);
            f32x4 g3 = *(const f32x4*)(p1 + 20);
            AFrag aT0, aT1;
            pack8(aT0, f0, f1, f2, f3);
            pack8(aT1, g0, g1, g2, g3);
            mm_step2(aT0, aT1, c * 32, rowOff, sW[0], sW[1], sig0, sig1, sp0, sp1);
        }
        // ---- K chunks 4..5: edge_attr (streamed once -> non-temporal) ----
#pragma unroll 1
        for (int c = 4; c < 6; ++c) {
            const float* p0 = a0e + (c & 1) * 32;
            const float* p1 = a1e + (c & 1) * 32;
            f32x4 f0 = __builtin_nontemporal_load((const f32x4*)(p0));
            f32x4 f1 = __builtin_nontemporal_load((const f32x4*)(p0 + 4));
            f32x4 f2 = __builtin_nontemporal_load((const f32x4*)(p0 + 16));
            f32x4 f3 = __builtin_nontemporal_load((const f32x4*)(p0 + 20));
            f32x4 g0 = __builtin_nontemporal_load((const f32x4*)(p1));
            f32x4 g1 = __builtin_nontemporal_load((const f32x4*)(p1 + 4));
            f32x4 g2 = __builtin_nontemporal_load((const f32x4*)(p1 + 16));
            f32x4 g3 = __builtin_nontemporal_load((const f32x4*)(p1 + 20));
            AFrag aT0, aT1;
            pack8(aT0, f0, f1, f2, f3);
            pack8(aT1, g0, g1, g2, g3);
            mm_step2(aT0, aT1, c * 32, rowOff, sW[0], sW[1], sig0, sig1, sp0, sp1);
        }

        // ---- activations + atomic scatter ----
        // C/D layout: lane column n = 16t+lo16 ; VGPR r holds row M = r + 8*(lane>>4)
#pragma unroll
        for (int half = 0; half < 2; ++half) {
            const v8f* as = half ? sig1 : sig0;
            const v8f* ap = half ? sp1  : sp0;
            const int dI  = half ? dI1  : dI0;
#pragma unroll
            for (int t = 0; t < 4; ++t) {
                const int col = t * 16 + lo16;
                float vals[8];
#pragma unroll
                for (int r = 0; r < 8; ++r) {
                    float xs = as[t][r] + bsv[t];
                    float g  = __builtin_amdgcn_rcpf(1.0f + __expf(-xs));
                    float xp = ap[t][r] + bpv[t];
                    float sv = (xp > 20.0f) ? xp : __logf(1.0f + __expf(xp));
                    vals[r] = g * sv;
                }
#pragma unroll
                for (int r = 0; r < 8; ++r) {
                    const int m  = r + half8;            // edge row within tile
                    const int dn = __shfl(dI, m, 32);    // lane m holds the row's dst
                    unsafeAtomicAdd(&agg[(size_t)dn * HID + col], vals[r]);
                }
            }
        }
    }
}

// ---------------------------------------------------------------------------
// Scalar fallback for E % 32 edges (not hit for the reference sizes).
// ---------------------------------------------------------------------------
__global__ void cgcnn_edge_tail_kernel(
    const float* __restrict__ x, const float* __restrict__ ea,
    const int* __restrict__ src, const int* __restrict__ dst,
    const float* __restrict__ Wsig, const float* __restrict__ bsig,
    const float* __restrict__ Wsp,  const float* __restrict__ bsp,
    float* __restrict__ agg, int eStart, int E)
{
    int e = eStart + blockIdx.x * blockDim.x + threadIdx.x;
    if (e >= E) return;
    const float* xs = x  + (size_t)src[e] * HID;
    const float* xd = x  + (size_t)dst[e] * HID;
    const float* ep = ea + (size_t)e * HID;
    const int d = dst[e];
    for (int c = 0; c < HID; ++c) {
        float s1 = bsig[c], s2 = bsp[c];
        const float* w1 = Wsig + c * ZDIM;
        const float* w2 = Wsp  + c * ZDIM;
        for (int k = 0; k < HID; ++k) {
            s1 += xs[k] * w1[k] + xd[k] * w1[64 + k] + ep[k] * w1[128 + k];
            s2 += xs[k] * w2[k] + xd[k] * w2[64 + k] + ep[k] * w2[128 + k];
        }
        float g  = 1.0f / (1.0f + __expf(-s1));
        float sp = (s2 > 20.0f) ? s2 : __logf(1.0f + __expf(s2));
        unsafeAtomicAdd(&agg[(size_t)d * HID + c], g * sp);
    }
}

// ---------------------------------------------------------------------------
// BN pass 1: h = x + agg -> out (temp), per-channel sum & sumsq reduction.
// gridDim*blockDim is a multiple of 64 so each thread owns one channel.
// ---------------------------------------------------------------------------
__global__ __launch_bounds__(256) void bn_stats_kernel(
    const float* __restrict__ x, const float* __restrict__ agg,
    float* __restrict__ hOut, float* __restrict__ sums, int total)
{
    __shared__ float ssum[HID];
    __shared__ float ssq[HID];
    if (threadIdx.x < HID) { ssum[threadIdx.x] = 0.0f; ssq[threadIdx.x] = 0.0f; }
    __syncthreads();

    const int tid    = blockIdx.x * blockDim.x + threadIdx.x;
    const int stride = blockDim.x * gridDim.x;
    float ls = 0.0f, lq = 0.0f;
    for (int i = tid; i < total; i += stride) {
        float h = x[i] + agg[i];
        hOut[i] = h;
        ls += h;
        lq += h * h;
    }
    const int ch = tid & (HID - 1);
    atomicAdd(&ssum[ch], ls);   // ds_add_f32
    atomicAdd(&ssq[ch],  lq);
    __syncthreads();
    if (threadIdx.x < HID) {
        unsafeAtomicAdd(&sums[threadIdx.x],       ssum[threadIdx.x]);
        unsafeAtomicAdd(&sums[HID + threadIdx.x], ssq[threadIdx.x]);
    }
}

// BN pass 2: fold mean/var/gamma/beta into per-channel scale & shift.
__global__ void bn_finalize_kernel(
    const float* __restrict__ sums, const float* __restrict__ gamma,
    const float* __restrict__ beta, float* __restrict__ ss, float invN)
{
    int c = threadIdx.x;
    if (c >= HID) return;
    float mean = sums[c] * invN;
    float var  = sums[HID + c] * invN - mean * mean;
    float sc   = gamma[c] * rsqrtf(var + 1e-5f);
    ss[c]        = sc;
    ss[HID + c]  = beta[c] - mean * sc;
}

// BN pass 3: out = h*scale + shift
__global__ __launch_bounds__(256) void bn_apply_kernel(
    float* __restrict__ out, const float* __restrict__ ss, int total)
{
    const int tid    = blockIdx.x * blockDim.x + threadIdx.x;
    const int stride = blockDim.x * gridDim.x;
    for (int i = tid; i < total; i += stride) {
        int c = i & (HID - 1);
        out[i] = out[i] * ss[c] + ss[HID + c];
    }
}

extern "C" void kernel_launch(void* const* d_in, const int* in_sizes, int n_in,
                              void* d_out, int out_size, void* d_ws, size_t ws_size,
                              hipStream_t stream)
{
    (void)n_in; (void)out_size; (void)ws_size;
    const float* x     = (const float*)d_in[0];
    const float* ea    = (const float*)d_in[1];
    const int*   src   = (const int*)  d_in[2];
    const int*   dst   = (const int*)  d_in[3];
    const float* Wsig  = (const float*)d_in[4];
    const float* bsig  = (const float*)d_in[5];
    const float* Wsp   = (const float*)d_in[6];
    const float* bsp   = (const float*)d_in[7];
    const float* gamma = (const float*)d_in[8];
    const float* beta  = (const float*)d_in[9];
    float* out = (float*)d_out;

    const int N = in_sizes[0] / HID;
    const int E = in_sizes[2];
    const size_t aggElems = (size_t)N * HID;

    float* agg  = (float*)d_ws;          // [N*64] scatter target
    float* sums = agg + aggElems;        // [128] sum | sumsq
    float* ss   = sums + 128;            // [128] scale | shift

    // zero agg + stat accumulators (graph-capturable memset node)
    hipMemsetAsync(d_ws, 0, (aggElems + 128) * sizeof(float), stream);

    const int nPairs = E >> 5;           // 32 edges per wave iteration
    if (nPairs > 0)
        cgcnn_edge_kernel<<<768, 256, 0, stream>>>(
            x, ea, src, dst, Wsig, bsig, Wsp, bsp, agg, nPairs);
    const int tailStart = nPairs << 5;
    if (E - tailStart > 0)
        cgcnn_edge_tail_kernel<<<1, 64, 0, stream>>>(
            x, ea, src, dst, Wsig, bsig, Wsp, bsp, agg, tailStart, E);

    const int total = (int)aggElems;
    bn_stats_kernel<<<1024, 256, 0, stream>>>(x, agg, out, sums, total);
    bn_finalize_kernel<<<1, 64, 0, stream>>>(sums, gamma, beta, ss, 1.0f / (float)N);
    bn_apply_kernel<<<1024, 256, 0, stream>>>(out, ss, total);
}
